// GridRNNSkew_69956427317706
// MI455X (gfx1250) — matrix operand order repaired and checked by reading, not compiled, not run on hardware
//
#include <hip/hip_runtime.h>

#define B_ 8
#define S_ 96
#define T_ 96
#define F_ 64
#define H_ 64
#define DEPTH_ 2

typedef __attribute__((ext_vector_type(16))) __bf16 v16bf;
typedef __attribute__((ext_vector_type(8)))  float  v8f;

// ---- WMMA fragment loaders (layouts per CDNA5 ISA 7.12.2, 16-bit A 16x32 / B 32x16) ----
__device__ __forceinline__ v16bf load_a_frag(const __bf16* row, int kblockBase, int laneHi) {
  v16bf r;
  uint4* p = reinterpret_cast<uint4*>(&r);
  const int base = kblockBase + laneHi * 8;
  p[0] = *reinterpret_cast<const uint4*>(row + base);
  p[1] = *reinterpret_cast<const uint4*>(row + base + 16);
  return r;
}
__device__ __forceinline__ v16bf load_b_frag(const __bf16* row, int kblockBase, int laneHi) {
  v16bf r;
  uint4* p = reinterpret_cast<uint4*>(&r);
  const int base = kblockBase + laneHi * 8;
  p[0] = *reinterpret_cast<const uint4*>(row + base);
  p[1] = *reinterpret_cast<const uint4*>(row + base + 16);
  return r;
}

// Single-instruction hardware tanh on gfx1250 (V_TANH_F32, TRANS32); branch-free fallback.
__device__ __forceinline__ float fast_tanh(float x) {
#if __has_builtin(__builtin_amdgcn_tanhf)
  return __builtin_amdgcn_tanhf(x);
#else
  const float e = __builtin_amdgcn_exp2f(x * 2.8853900817779268f);
  return 1.0f - 2.0f * __builtin_amdgcn_rcpf(e + 1.0f);
#endif
}

// Convert 8 contiguous f32 -> 8 bf16 packed in a uint4 (16B).
__device__ __forceinline__ uint4 cvt8_f32_bf16(const float* __restrict__ p) {
  const float4 a = *reinterpret_cast<const float4*>(p);
  const float4 b = *reinterpret_cast<const float4*>(p + 4);
  union { __bf16 h[8]; uint4 u; } r;
  r.h[0] = (__bf16)a.x; r.h[1] = (__bf16)a.y; r.h[2] = (__bf16)a.z; r.h[3] = (__bf16)a.w;
  r.h[4] = (__bf16)b.x; r.h[5] = (__bf16)b.y; r.h[6] = (__bf16)b.z; r.h[7] = (__bf16)b.w;
  return r.u;
}

// ---- CDNA5 async DMA: global -> LDS, 16B per lane, tracked by ASYNCcnt (ISA 15.18.3) ----
// VDST = per-lane LDS byte address, VADDR = 64-bit global address (GV mode).
__device__ __forceinline__ void async_load_b128(unsigned lds_off, const float* gaddr) {
  asm volatile("global_load_async_to_lds_b128 %0, %1, off"
               :: "v"(lds_off), "v"((unsigned long long)(size_t)gaddr)
               : "memory");
}
__device__ __forceinline__ void wait_asynccnt0() {
#if __has_builtin(__builtin_amdgcn_s_wait_asynccnt)
  __builtin_amdgcn_s_wait_asynccnt(0);
#else
  asm volatile("s_wait_asynccnt 0x0" ::: "memory");
#endif
}

// ---------------- LDS layout (worst-case union of both paths; dynamic) ----------------
#define SM_WT    0                        // bf16 weights: SPLIT [128n][128k] / FUSED [128n][256k]
#define SM_AST   65536                    // bf16 A stage: SPLIT [96c][128k]  / FUSED [96c][256k]
#define SM_HX    114688                   // bf16 hx state [96 j][64]
#define SM_HY    126976                   // bf16 hy state [96 i][64]
#define SM_BUF4  139264                   // SPLIT: f32 pxy_s x2 buffers [96c][128n]; FUSED: bf16 px0/py0
#define SM_PX0F  237568                   // SPLIT: f32 px0f [96][64]
#define SM_PY0F  262144                   // SPLIT: f32 py0f [96][64]
#define SM_WAVE_BYTES 286720              // 280 KB (<= 320 KB per WGP)
#define PXYN (96 * 128)                   // floats per pxy buffer

// One workgroup (8 wave32) per batch; wavefront over 191 anti-diagonals.
// SPLIT=true : h = tanh( pxy(i,j) + [hx(i-1,j)|hy(i,j-1)] @ [Whx|Why] )  K=128, 4 WMMA,
//              depth-1 pxy prefetched one diagonal ahead via async-to-LDS DMA (double buffer).
// SPLIT=false: input projections folded via identity blocks, K=256, 8 WMMA (no-workspace fallback).
template <bool SPLIT>
__global__ void __launch_bounds__(256) grid_rnn_wavefront(
    const float* __restrict__ src, const float* __restrict__ trg,
    const float* __restrict__ Wi_x, const float* __restrict__ Wh_x, const float* __restrict__ b_x,
    const float* __restrict__ Wi_y, const float* __restrict__ Wh_y, const float* __restrict__ b_y,
    float* __restrict__ out, const float* __restrict__ pxy_g, int depth)
{
  constexpr int KB  = SPLIT ? 4 : 8;      // 32-wide K blocks
  constexpr int KW  = SPLIT ? 128 : 256;  // A / W_T row width
  constexpr int KSH = SPLIT ? 7 : 8;
  constexpr int NDIAG = S_ + T_ - 1;

  extern __shared__ char smem[];
  __bf16* W_T     = (__bf16*)(smem + SM_WT);
  __bf16* A_stage = (__bf16*)(smem + SM_AST);
  __bf16* hx_col  = (__bf16*)(smem + SM_HX);
  __bf16* hy_row  = (__bf16*)(smem + SM_HY);
  float*  pxy_s   = (float*)(smem + SM_BUF4);            // SPLIT (x2 buffers)
  __bf16* px0_s   = (__bf16*)(smem + SM_BUF4);           // FUSED
  __bf16* py0_s   = px0_s + 96 * 64;                     // FUSED
  float*  px0f    = (float*)(smem + SM_PX0F);            // SPLIT depth 0
  float*  py0f    = (float*)(smem + SM_PY0F);            // SPLIT depth 0

  const int tid    = threadIdx.x;
  const int b      = blockIdx.x;
  const int lane   = tid & 31;
  const int laneLo = lane & 15;
  const int laneHi = lane >> 4;
  const int wave   = tid >> 5;              // wave w owns N-tile w (0..7)

  // ---------- Build transposed fused weight matrix W_T[n][k] (bf16) ----------
  const float* Whx  = Wh_x + (size_t)depth * 2 * H_ * H_;
  const float* Why  = Wh_y + (size_t)depth * 2 * H_ * H_;
  const float* Wix1 = Wi_x + (size_t)F_ * H_;
  const float* Wiy1 = Wi_y + (size_t)F_ * H_;
  for (int idx = tid; idx < 128 * KW; idx += 256) {
    const int n = idx >> KSH, k = idx & (KW - 1);
    float w;
    if (k < 128) {
      w = (n < 64) ? Whx[k * H_ + n] : Why[k * H_ + (n - 64)];
    } else if (k < 192) {                 // FUSED only
      const int r = k - 128;
      w = (depth == 0) ? ((n == r) ? 1.f : 0.f)
                       : ((n < 64) ? Wix1[r * H_ + n] : 0.f);
    } else {
      const int r = k - 192;
      w = (depth == 0) ? ((n == 64 + r) ? 1.f : 0.f)
                       : ((n >= 64) ? Wiy1[r * H_ + (n - 64)] : 0.f);
    }
    W_T[n * KW + k] = (__bf16)w;
  }

  // ---------- Depth-0 input projections (SPLIT: f32 + bias folded; FUSED: bf16) ----------
  if (depth == 0) {
    for (int idx = tid; idx < S_ * H_; idx += 256) {
      const int i = idx >> 6, h = idx & 63;
      const float* srow = src + ((size_t)b * S_ + i) * F_;
      float s = SPLIT ? b_x[h] : 0.f;
#pragma unroll 8
      for (int f = 0; f < F_; ++f) s += srow[f] * Wi_x[f * H_ + h];
      if constexpr (SPLIT) px0f[idx] = s; else px0_s[idx] = (__bf16)s;
    }
    for (int idx = tid; idx < T_ * H_; idx += 256) {
      const int j = idx >> 6, h = idx & 63;
      const float* trow = trg + ((size_t)b * T_ + j) * F_;
      float s = SPLIT ? b_y[h] : 0.f;
#pragma unroll 8
      for (int f = 0; f < F_; ++f) s += trow[f] * Wi_y[f * H_ + h];
      if constexpr (SPLIT) py0f[idx] = s; else py0_s[idx] = (__bf16)s;
    }
  }

  const float* pxyRowBase = pxy_g + (size_t)b * S_ * T_ * 128;

  // SPLIT depth-1: pre-queue async DMA of diagonal 0's pxy into buffer 0.
  if constexpr (SPLIT) {
    if (depth != 0) {
      // diag 0: single cell (0,0)
      for (int idx = tid; idx < 32; idx += 256) {
        const int n0 = (idx & 31) << 2;
        const unsigned loff = (unsigned)((const char*)(pxy_s + n0) - smem);
        async_load_b128(loff, pxyRowBase + n0);
      }
    }
  }
  __syncthreads();

  // ---------- Hoist weight-stationary B fragments (live in VGPRs for all 191 steps) ----------
  const int ncol = wave * 16 + laneLo;            // global N column 0..127
  v16bf Bfr[KB];
  {
    const __bf16* brow = W_T + (size_t)ncol * KW;
#pragma unroll
    for (int kb = 0; kb < KB; ++kb) Bfr[kb] = load_b_frag(brow, kb * 32, laneHi);
  }
  const float biasv = (ncol < 64) ? b_x[depth * H_ + ncol] : b_y[depth * H_ + (ncol - 64)];
  const int   sel   = (ncol >= 64);               // wave-uniform: 0 -> hx, 1 -> hy
  const int   nloc  = ncol & 63;
  __bf16*     statePtr = sel ? hy_row : hx_col;   // hx keyed by j, hy keyed by i

  const size_t STH = (size_t)S_ * T_ * H_;
  float*       outBase = out + ((size_t)(sel * DEPTH_ + depth) * B_ + b) * STH;
  const float* gx = out + (size_t)b * STH;                           // hx depth0 (unmasked)
  const float* gy = out + ((size_t)(1 * DEPTH_ + 0) * B_ + b) * STH; // hy depth0 (unmasked)

  const uint4 zero4 = {0u, 0u, 0u, 0u};

  // ---------- Wavefront over anti-diagonals ----------
  for (int d = 0; d < NDIAG; ++d) {
    const int iLo = (d > T_ - 1) ? (d - (T_ - 1)) : 0;
    const int iHi = (d < S_ - 1) ? d : (S_ - 1);
    const int cnt = iHi - iLo + 1;                 // cells on this diagonal (uniform)

    if constexpr (SPLIT) {
      // Stage recurrent state (K 0..127), 8 bf16 per iteration.
      for (int idx = tid; idx < (cnt << 4); idx += 256) {
        const int c = idx >> 4, k0 = (idx & 15) << 3;
        const int i = iLo + c, j = d - i;
        uint4 v;
        if (k0 < 64) v = (i == 0) ? zero4 : *reinterpret_cast<const uint4*>(hx_col + j * 64 + k0);
        else         v = (j == 0) ? zero4 : *reinterpret_cast<const uint4*>(hy_row + i * 64 + (k0 - 64));
        *reinterpret_cast<uint4*>(A_stage + (c << 7) + k0) = v;
      }
      if (depth == 0) {
        // Synchronous pxy staging from LDS projections (nothing to hide).
        float* dst = pxy_s + (d & 1) * PXYN;
        for (int idx = tid; idx < (cnt << 5); idx += 256) {
          const int c = idx >> 5, n0 = (idx & 31) << 2;
          const int i = iLo + c, j = d - i;
          const float4 v = (n0 < 64) ? *reinterpret_cast<const float4*>(px0f + i * 64 + n0)
                                     : *reinterpret_cast<const float4*>(py0f + j * 64 + (n0 - 64));
          *reinterpret_cast<float4*>(dst + (c << 7) + n0) = v;
        }
      } else {
        // Current diagonal's pxy was queued last iteration; ensure it has landed,
        // then queue next diagonal's pxy into the other buffer (overlaps with compute).
        wait_asynccnt0();
        const int dn = d + 1;
        if (dn < NDIAG) {
          const int iLo1 = (dn > T_ - 1) ? (dn - (T_ - 1)) : 0;
          const int iHi1 = (dn < S_ - 1) ? dn : (S_ - 1);
          const int cnt1 = iHi1 - iLo1 + 1;
          float* dst = pxy_s + (dn & 1) * PXYN;
          for (int idx = tid; idx < (cnt1 << 5); idx += 256) {
            const int c = idx >> 5, n0 = (idx & 31) << 2;
            const int i = iLo1 + c, j = dn - i;
            const unsigned loff = (unsigned)((const char*)(dst + (c << 7) + n0) - smem);
            async_load_b128(loff, pxyRowBase + ((size_t)i * T_ + j) * 128 + n0);
          }
        }
      }
    } else {
      for (int idx = tid; idx < (cnt << 5); idx += 256) {
        const int c  = idx >> 5;
        const int k0 = (idx & 31) << 3;
        const int i  = iLo + c, j = d - i;
        uint4 v;
        if (k0 < 64) {
          v = (i == 0) ? zero4 : *reinterpret_cast<const uint4*>(hx_col + j * 64 + k0);
        } else if (k0 < 128) {
          v = (j == 0) ? zero4 : *reinterpret_cast<const uint4*>(hy_row + i * 64 + (k0 - 64));
        } else if (k0 < 192) {
          v = (depth == 0) ? *reinterpret_cast<const uint4*>(px0_s + i * 64 + (k0 - 128))
                           : cvt8_f32_bf16(gx + ((size_t)i * T_ + j) * H_ + (k0 - 128));
        } else {
          v = (depth == 0) ? *reinterpret_cast<const uint4*>(py0_s + j * 64 + (k0 - 192))
                           : cvt8_f32_bf16(gy + ((size_t)i * T_ + j) * H_ + (k0 - 192));
        }
        *reinterpret_cast<uint4*>(A_stage + (c << 8) + k0) = v;
      }
    }
    __syncthreads();   // all reads of old state done; staged data visible to all waves

    const float* pxyCur = pxy_s + (d & 1) * PXYN;   // SPLIT only
    const int mtiles = (cnt + 15) >> 4;
    for (int mt = 0; mt < mtiles; ++mt) {
      const __bf16* arow = A_stage + (size_t)(mt * 16 + laneLo) * KW;
      v16bf Afr[KB];
#pragma unroll
      for (int kb = 0; kb < KB; ++kb) Afr[kb] = load_a_frag(arow, kb * 32, laneHi);

      v8f acc;
      if constexpr (SPLIT) {
        const float* ccol = pxyCur + ncol;          // C layout: VGPR r -> row mt*16+r+8*laneHi
#pragma unroll
        for (int r = 0; r < 8; ++r) acc[r] = ccol[(mt * 16 + r + laneHi * 8) << 7];
      } else {
        acc = (v8f){biasv, biasv, biasv, biasv, biasv, biasv, biasv, biasv};
      }
#pragma unroll
      for (int kb = 0; kb < KB; ++kb) {
        acc = __builtin_amdgcn_wmma_f32_16x16x32_bf16(
            false, Afr[kb], false, Bfr[kb], (short)0, acc, false, false);
      }

      float hval[8];
#pragma unroll
      for (int r = 0; r < 8; ++r) hval[r] = fast_tanh(acc[r]);

#pragma unroll
      for (int r = 0; r < 8; ++r) {
        const int c = mt * 16 + r + laneHi * 8;
        if (c < cnt) {
          const int i = iLo + c, j = d - i;
          outBase[((size_t)i * T_ + j) * H_ + nloc] = hval[r];   // unmasked output
          const int sidx = sel ? i : j;
          statePtr[sidx * 64 + nloc] = (__bf16)hval[r];          // state for next diagonal
        }
      }
    }
    __syncthreads();   // state writes visible before next diagonal's staging
  }
}

// Fully parallel pre-pass (SPLIT path, 576 WGs): for every cell,
//   pxy[flat][0:64]   = hx_d0(b,i,j) @ Wi_x[1] + b_x[1]
//   pxy[flat][64:128] = hy_d0(b,i,j) @ Wi_y[1] + b_y[1]
__global__ void __launch_bounds__(256) grid_rnn_project(
    const float* __restrict__ Wi_x, const float* __restrict__ Wi_y,
    const float* __restrict__ b_x,  const float* __restrict__ b_y,
    const float* __restrict__ out, float* __restrict__ pxy_g)
{
  extern __shared__ char smem[];
  __bf16* Wp_T = (__bf16*)smem;              // [128 n][128 k] 32 KB
  __bf16* Ap   = (__bf16*)(smem + 32768);    // [128 c][128 k] 32 KB

  const int tid = threadIdx.x, lane = tid & 31;
  const int laneLo = lane & 15, laneHi = lane >> 4, wave = tid >> 5;
  const float* Wix1 = Wi_x + (size_t)F_ * H_;
  const float* Wiy1 = Wi_y + (size_t)F_ * H_;

  for (int idx = tid; idx < 128 * 128; idx += 256) {
    const int n = idx >> 7, k = idx & 127;
    const float w = (k < 64) ? ((n < 64) ? Wix1[k * H_ + n] : 0.f)
                             : ((n >= 64) ? Wiy1[(k - 64) * H_ + (n - 64)] : 0.f);
    Wp_T[n * 128 + k] = (__bf16)w;
  }

  const size_t cell0  = (size_t)blockIdx.x * 128;            // 73728 cells / 128 per WG
  const float* gx_all = out;                                 // hx depth0, [flat][64]
  const float* gy_all = out + (size_t)2 * B_ * S_ * T_ * H_; // hy depth0
  for (int idx = tid; idx < 128 * 16; idx += 256) {
    const int c = idx >> 4, k0 = (idx & 15) << 3;
    const size_t flat = cell0 + c;
    const uint4 v = (k0 < 64) ? cvt8_f32_bf16(gx_all + flat * 64 + k0)
                              : cvt8_f32_bf16(gy_all + flat * 64 + (k0 - 64));
    *reinterpret_cast<uint4*>(Ap + (c << 7) + k0) = v;
  }
  __syncthreads();

  const int ncol = wave * 16 + laneLo;
  v16bf Bf[4];
  {
    const __bf16* brow = Wp_T + (size_t)ncol * 128;
#pragma unroll
    for (int kb = 0; kb < 4; ++kb) Bf[kb] = load_b_frag(brow, kb * 32, laneHi);
  }
  const float biasv = (ncol < 64) ? b_x[H_ + ncol] : b_y[H_ + (ncol - 64)];  // depth-1 biases

  for (int mt = 0; mt < 8; ++mt) {
    const __bf16* arow = Ap + (size_t)(mt * 16 + laneLo) * 128;
    v16bf Afr[4];
#pragma unroll
    for (int kb = 0; kb < 4; ++kb) Afr[kb] = load_a_frag(arow, kb * 32, laneHi);
    v8f acc = {biasv, biasv, biasv, biasv, biasv, biasv, biasv, biasv};
#pragma unroll
    for (int kb = 0; kb < 4; ++kb)
      acc = __builtin_amdgcn_wmma_f32_16x16x32_bf16(false, Afr[kb], false, Bf[kb],
                                                    (short)0, acc, false, false);
#pragma unroll
    for (int r = 0; r < 8; ++r) {
      const int c = mt * 16 + r + laneHi * 8;
      pxy_g[(cell0 + c) * 128 + ncol] = acc[r];
    }
  }
}

// Final pass: zero cells outside each sample's (src_len x trg_len) grid.
__global__ void grid_rnn_mask(float* __restrict__ out,
                              const int* __restrict__ src_lens,
                              const int* __restrict__ trg_lens)
{
  const size_t total = (size_t)2 * DEPTH_ * B_ * S_ * T_ * H_;
  size_t idx = (size_t)blockIdx.x * blockDim.x + threadIdx.x;
  if (idx >= total) return;
  const size_t per = (size_t)S_ * T_ * H_;
  const int b = (int)((idx / per) % B_);
  const int i = (int)((idx / ((size_t)T_ * H_)) % S_);
  const int j = (int)((idx / H_) % T_);
  if (i >= src_lens[b] || j >= trg_lens[b]) out[idx] = 0.f;
}

extern "C" void kernel_launch(void* const* d_in, const int* in_sizes, int n_in,
                              void* d_out, int out_size, void* d_ws, size_t ws_size,
                              hipStream_t stream) {
  (void)in_sizes; (void)n_in; (void)out_size;
  const float* src  = (const float*)d_in[0];
  const float* trg  = (const float*)d_in[1];
  const float* Wi_x = (const float*)d_in[2];
  const float* Wh_x = (const float*)d_in[3];
  const float* b_x  = (const float*)d_in[4];
  const float* Wi_y = (const float*)d_in[5];
  const float* Wh_y = (const float*)d_in[6];
  const float* b_y  = (const float*)d_in[7];
  const int* src_lens = (const int*)d_in[8];
  const int* trg_lens = (const int*)d_in[9];
  float* out = (float*)d_out;

  const size_t pxyBytes = (size_t)B_ * S_ * T_ * 128 * sizeof(float);  // 37.75 MB

  if (ws_size >= pxyBytes) {
    float* pxy = (float*)d_ws;
    grid_rnn_wavefront<true><<<B_, 256, SM_WAVE_BYTES, stream>>>(
        src, trg, Wi_x, Wh_x, b_x, Wi_y, Wh_y, b_y, out, pxy, 0);
    grid_rnn_project<<<(B_ * S_ * T_) / 128, 256, 65536, stream>>>(
        Wi_x, Wi_y, b_x, b_y, out, pxy);
    grid_rnn_wavefront<true><<<B_, 256, SM_WAVE_BYTES, stream>>>(
        src, trg, Wi_x, Wh_x, b_x, Wi_y, Wh_y, b_y, out, pxy, 1);
  } else {
    grid_rnn_wavefront<false><<<B_, 256, SM_WAVE_BYTES, stream>>>(
        src, trg, Wi_x, Wh_x, b_x, Wi_y, Wh_y, b_y, out, nullptr, 0);
    grid_rnn_wavefront<false><<<B_, 256, SM_WAVE_BYTES, stream>>>(
        src, trg, Wi_x, Wh_x, b_x, Wi_y, Wh_y, b_y, out, nullptr, 1);
  }

  const size_t total = (size_t)2 * DEPTH_ * B_ * S_ * T_ * H_;
  const int blocks = (int)((total + 255) / 256);
  grid_rnn_mask<<<blocks, 256, 0, stream>>>(out, src_lens, trg_lens);
}
